// LiZAttention_26860725469326
// MI455X (gfx1250) — compile-verified, matrix operand order checked by reading, not run on hardware
//
#include <hip/hip_runtime.h>
#include <hip/hip_bf16.h>
#include <cstdint>
#include <cstddef>

// ---------------------------------------------------------------------------
// LiZAttention for MI455X (gfx1250, wave32, WMMA).
// Sizes: B=2, N=1024, HID=2048, H=32, D=64, HKV=8, GROUPS=4, CHUNK=64.
// Heavy matmuls: v_wmma_f32_16x16x32_bf16, bf16 tiles staged in LDS so each
// lane's fragment is two ds_load_b128 (per the CDNA5 16-bit A/B VGPR layout).
// q/k chunk tiles in the DeltaNet kernel use GLOBAL_LOAD_ASYNC_TO_LDS_B128.
// ---------------------------------------------------------------------------

typedef __attribute__((ext_vector_type(16))) __bf16 v16bf;
typedef __attribute__((ext_vector_type(8)))  __bf16 v8bf;
typedef __attribute__((ext_vector_type(2)))  __bf16 v2bf;
typedef __attribute__((ext_vector_type(8)))  float  v8f;

#define BB   2
#define NN   1024
#define HIDC 2048
#define HH   32
#define DD   64
#define HKVC 8
#define NEGINF (-3.4028235e38f)

// Optional gfx1250 async global->LDS copy path (ASYNCcnt).
#if defined(__has_builtin)
#  if __has_builtin(__builtin_amdgcn_global_load_async_to_lds_b128) && \
      __has_builtin(__builtin_amdgcn_s_wait_asynccnt)
#    define USE_ASYNC 1
#  endif
#endif
#ifndef USE_ASYNC
#  define USE_ASYNC 0
#endif

#if USE_ASYNC
// Probe-learned signature: both pointers are generic `int __vector(4)*`.
typedef int async_v4i __attribute__((vector_size(16)));
#define ASYNC_PTR(p) ((async_v4i*)(p))
#endif

// K-index of fragment element e for lane-half hf (CDNA5 16-bit A/B layout):
// half 0: e0..7 -> K0..7,  e8..15 -> K16..23
// half 1: e0..7 -> K8..15, e8..15 -> K24..31
__device__ __forceinline__ int kmap(int e, int hf) {
  return (e < 8) ? (hf * 8 + e) : (8 + hf * 8 + e);
}

__device__ __forceinline__ v8f wmma_bf16(v16bf a, v16bf b, v8f c) {
  return __builtin_amdgcn_wmma_f32_16x16x32_bf16(
      /*neg_a=*/false, a, /*neg_b=*/false, b,
      /*c_mod=*/(short)0, c, /*reuse_a=*/false, /*reuse_b=*/false);
}

// Build a 16-element fragment from two contiguous 8-element LDS runs.
__device__ __forceinline__ v16bf frag_from_lds(const __bf16* row, int hf) {
  v8bf lo = *(const v8bf*)(row + hf * 8);        // K = hf*8 .. +7
  v8bf hi = *(const v8bf*)(row + 16 + hf * 8);   // K = 16+hf*8 .. +7
  return __builtin_shufflevector(lo, hi, 0, 1, 2, 3, 4, 5, 6, 7,
                                 8, 9, 10, 11, 12, 13, 14, 15);
}

// ---------------------------------------------------------------------------
// Generic LDS-staged WMMA GEMM tile core.
// Block = 128 threads = 4 waves computing a 16(M) x 64(N) slab of
//   C = alpha * A*op(B) + beta * C   (op = transpose if TRANSB)
// A: fp32 [M,K] row-major (lda). B: fp32 [K,N] (ldb) or [N,K] if TRANSB.
// Tiles staged as bf16: sA row-major [16][40], sB column-major [64][40]
// (row stride 80 B keeps every 8-element run 16 B aligned, conflict-free).
// causal: entries with col > row get -FLT_MAX (pre-softmax scores).
// M,N,K multiples of 16/64/32 for every use below.
// ---------------------------------------------------------------------------
template <bool TRANSB>
__device__ __forceinline__ void gemm_core(const float* __restrict__ A, int lda,
                                          const float* __restrict__ Bm, int ldb,
                                          float* __restrict__ C, int ldc,
                                          int K, float alpha, float beta,
                                          bool causal) {
  __shared__ __attribute__((aligned(16))) __bf16 sA[16][40];
  __shared__ __attribute__((aligned(16))) __bf16 sB[64][40];  // sB[n][k]

  const int tid  = threadIdx.x;
  const int wave = tid >> 5;
  const int lane = tid & 31;
  const int hf   = lane >> 4;
  const int idx  = lane & 15;

  const int row0     = blockIdx.y * 16;
  const int colBlock = blockIdx.x * 64;
  const int col0     = colBlock + wave * 16;

  if (causal && colBlock > row0 + 15) {  // fully masked tile
    for (int r = 0; r < 8; ++r) {
      int rg = row0 + hf * 8 + r;
      C[(size_t)rg * ldc + col0 + idx] = NEGINF;
    }
    return;
  }

  v8f acc = {0.f, 0.f, 0.f, 0.f, 0.f, 0.f, 0.f, 0.f};

  for (int k0 = 0; k0 < K; k0 += 32) {
    // ---- stage A tile: 16x32, pack fp32 pairs -> bf16 (coalesced b64 reads)
    for (int i = tid; i < 16 * 16; i += 128) {
      int r = i >> 4, kp = i & 15;
      float2 f = *(const float2*)&A[(size_t)(row0 + r) * lda + k0 + 2 * kp];
      v2bf t = {(__bf16)f.x, (__bf16)f.y};
      *(v2bf*)&sA[r][2 * kp] = t;
    }
    // ---- stage B tile (column-major in LDS: sB[n][k])
    if (!TRANSB) {
      for (int i = tid; i < 64 * 16; i += 128) {
        int n = i & 63, kp = i >> 6;  // consecutive tid -> consecutive n
        float f0 = Bm[(size_t)(k0 + 2 * kp) * ldb + colBlock + n];
        float f1 = Bm[(size_t)(k0 + 2 * kp + 1) * ldb + colBlock + n];
        v2bf t = {(__bf16)f0, (__bf16)f1};
        *(v2bf*)&sB[n][2 * kp] = t;
      }
    } else {
      for (int i = tid; i < 64 * 16; i += 128) {
        int kp = i & 15, n = i >> 4;  // consecutive tid -> contiguous K
        float2 f = *(const float2*)&Bm[(size_t)(colBlock + n) * ldb + k0 + 2 * kp];
        v2bf t = {(__bf16)f.x, (__bf16)f.y};
        *(v2bf*)&sB[n][2 * kp] = t;
      }
    }
    if (k0 + 32 < K) {  // prefetch next K-slab
      __builtin_prefetch(&A[(size_t)(row0 + (tid & 15)) * lda + k0 + 32], 0, 1);
      if (!TRANSB)
        __builtin_prefetch(&Bm[(size_t)(k0 + 32 + (tid & 31)) * ldb + colBlock], 0, 1);
      else
        __builtin_prefetch(&Bm[(size_t)(colBlock + (tid & 63)) * ldb + k0 + 32], 0, 1);
    }
    __syncthreads();

    v16bf a = frag_from_lds(&sA[idx][0], hf);
    v16bf b = frag_from_lds(&sB[wave * 16 + idx][0], hf);
    acc = wmma_bf16(a, b, acc);
    __syncthreads();
  }

  // ---- store C tile (lanes 0-15 -> rows r, lanes 16-31 -> rows 8+r)
#pragma unroll
  for (int r = 0; r < 8; ++r) {
    int rg = row0 + hf * 8 + r;
    int cg = col0 + idx;
    float v = alpha * acc[r];
    if (causal && cg > rg) {
      v = NEGINF;
    } else if (beta != 0.f) {
      v += beta * C[(size_t)rg * ldc + cg];
    }
    C[(size_t)rg * ldc + cg] = v;
  }
}

// ---------------- wrapper kernels over gemm_core ----------------

__global__ __launch_bounds__(128) void gemm_nn_kernel(
    const float* A, int lda, const float* Bm, int ldb, float* C, int ldc,
    int K, float alpha, float beta) {
  gemm_core<false>(A, lda, Bm, ldb, C, ldc, K, alpha, beta, false);
}

__global__ __launch_bounds__(128) void scores_kernel(
    const float* __restrict__ q, const float* __restrict__ kf,
    float* __restrict__ attn) {
  int bh = blockIdx.z;
  int b = bh >> 5, h = bh & 31, kvh = h >> 2;
  const float* A  = q  + (size_t)b * NN * HIDC + h * DD;
  const float* Bm = kf + (size_t)b * NN * (HKVC * DD) + kvh * DD;
  float* C = attn + (size_t)bh * NN * NN;
  gemm_core<true>(A, HIDC, Bm, HKVC * DD, C, NN, DD, 0.125f, 0.f, true);
}

__global__ __launch_bounds__(128) void obase_kernel(
    const float* __restrict__ attn, const float* __restrict__ vf,
    float* __restrict__ ocomb) {
  int bh = blockIdx.z;
  int b = bh >> 5, h = bh & 31, kvh = h >> 2;
  const float* A  = attn + (size_t)bh * NN * NN;
  const float* Bm = vf + (size_t)b * NN * (HKVC * DD) + kvh * DD;
  float* C = ocomb + (size_t)b * NN * HIDC + h * DD;
  gemm_core<false>(A, NN, Bm, HKVC * DD, C, HIDC, NN, 0.5f, 1.f, false);
}

// ---------------------------------------------------------------------------
// Row softmax over the attention matrix (in place). One block per row.
// ---------------------------------------------------------------------------
__global__ __launch_bounds__(256) void softmax_rows(float* __restrict__ attn) {
  __shared__ float redm[8];
  __shared__ float reds[8];
  float* p = attn + (size_t)blockIdx.x * NN;
  int tid = threadIdx.x;

  float m = NEGINF;
  for (int c = tid; c < NN; c += 256) m = fmaxf(m, p[c]);
  for (int o = 16; o; o >>= 1) m = fmaxf(m, __shfl_xor(m, o, 32));
  if ((tid & 31) == 0) redm[tid >> 5] = m;
  __syncthreads();
  float bm = redm[0];
#pragma unroll
  for (int i = 1; i < 8; ++i) bm = fmaxf(bm, redm[i]);

  float s = 0.f;
  for (int c = tid; c < NN; c += 256) {
    float e = __expf(p[c] - bm);
    p[c] = e;
    s += e;
  }
  for (int o = 16; o; o >>= 1) s += __shfl_xor(s, o, 32);
  if ((tid & 31) == 0) reds[tid >> 5] = s;
  __syncthreads();
  float tot = 0.f;
#pragma unroll
  for (int i = 0; i < 8; ++i) tot += reds[i];
  float inv = 1.0f / tot;
  for (int c = tid; c < NN; c += 256) p[c] *= inv;
}

// ---------------------------------------------------------------------------
// Feature maps: per 64-wide group, dst = softmax(src); optionally
// gate = log_sigmoid(src)/16. One wave per group (2 elems/lane, wave32).
// ---------------------------------------------------------------------------
__device__ __forceinline__ float log_sigmoid(float x) {
  return fminf(x, 0.f) - log1pf(__expf(-fabsf(x)));
}

__global__ __launch_bounds__(256) void featmap_kernel(
    const float* __restrict__ src, float* __restrict__ dst,
    float* __restrict__ gate) {
  int g = blockIdx.x * 8 + (threadIdx.x >> 5);
  int lane = threadIdx.x & 31;
  const float* s = src + (size_t)g * 64;
  float x0 = s[lane], x1 = s[lane + 32];
  float m = fmaxf(x0, x1);
  for (int o = 16; o; o >>= 1) m = fmaxf(m, __shfl_xor(m, o, 32));
  float e0 = __expf(x0 - m), e1 = __expf(x1 - m);
  float ss = e0 + e1;
  for (int o = 16; o; o >>= 1) ss += __shfl_xor(ss, o, 32);
  float inv = 1.0f / ss;
  dst[(size_t)g * 64 + lane]      = e0 * inv;
  dst[(size_t)g * 64 + lane + 32] = e1 * inv;
  if (gate != nullptr) {
    gate[(size_t)g * 64 + lane]      = log_sigmoid(x0) * 0.0625f;
    gate[(size_t)g * 64 + lane + 32] = log_sigmoid(x1) * 0.0625f;
  }
}

// ---------------------------------------------------------------------------
// 64x64x64 matmul on LDS-resident fp32 matrices via WMMA. Row stride 68
// floats (272 B, a 16 B multiple) so the contiguous operand runs load as
// ds_load_b128. Block = 256 threads = 8 waves, two 16x16 tiles per wave.
//   C = alpha * opA(A) * opB(B) + beta * C      (MASKM==1: zero where col>row)
// ---------------------------------------------------------------------------
__device__ __forceinline__ v16bf frag_contig_f32(const float* run, int k0, int hf) {
  const float* p = run + k0 + hf * 8;
  float4 r0 = *(const float4*)(p);
  float4 r1 = *(const float4*)(p + 4);
  float4 r2 = *(const float4*)(p + 16);
  float4 r3 = *(const float4*)(p + 20);
  v16bf f;
  f[0]=(__bf16)r0.x; f[1]=(__bf16)r0.y; f[2]=(__bf16)r0.z; f[3]=(__bf16)r0.w;
  f[4]=(__bf16)r1.x; f[5]=(__bf16)r1.y; f[6]=(__bf16)r1.z; f[7]=(__bf16)r1.w;
  f[8]=(__bf16)r2.x; f[9]=(__bf16)r2.y; f[10]=(__bf16)r2.z; f[11]=(__bf16)r2.w;
  f[12]=(__bf16)r3.x; f[13]=(__bf16)r3.y; f[14]=(__bf16)r3.z; f[15]=(__bf16)r3.w;
  return f;
}

template <bool TA, bool TB, int MASKM>
__device__ __forceinline__ void mm64(const float (*A)[68], const float (*B)[68],
                                     float (*C)[68], float alpha, float beta) {
  const int tid  = threadIdx.x;
  const int wave = tid >> 5;
  const int lane = tid & 31;
  const int hf   = lane >> 4;
  const int idx  = lane & 15;

#pragma unroll
  for (int tt = 0; tt < 2; ++tt) {
    int t = wave * 2 + tt;
    int trow = (t >> 2) * 16;
    int tcol = (t & 3) * 16;
    v8f acc = {0.f, 0.f, 0.f, 0.f, 0.f, 0.f, 0.f, 0.f};
#pragma unroll
    for (int k0 = 0; k0 < 64; k0 += 32) {
      v16bf a, b;
      if (!TA) {
        a = frag_contig_f32(&A[trow + idx][0], k0, hf);   // contiguous rows
      } else {
#pragma unroll
        for (int e = 0; e < 16; ++e)
          a[e] = (__bf16)A[k0 + kmap(e, hf)][trow + idx]; // strided gather
      }
      if (TB) {
        b = frag_contig_f32(&B[tcol + idx][0], k0, hf);   // contiguous rows
      } else {
#pragma unroll
        for (int e = 0; e < 16; ++e)
          b[e] = (__bf16)B[k0 + kmap(e, hf)][tcol + idx]; // strided gather
      }
      acc = wmma_bf16(a, b, acc);
    }
#pragma unroll
    for (int r = 0; r < 8; ++r) {
      int rg = trow + hf * 8 + r;
      int cg = tcol + idx;
      float v = alpha * acc[r];
      if (beta != 0.f) v += beta * C[rg][cg];
      if (MASKM == 1 && cg > rg) v = 0.f;
      C[rg][cg] = v;
    }
  }
}

// ---------------------------------------------------------------------------
// DeltaNet chunked recurrence. One 256-thread block per (b,h); sequentially
// scans 16 chunks of 64 keeping S[64x64] in LDS (~104 KB/group; 320 KB/WGP
// on CDNA5 -> 3 groups resident per WGP). Writes MAG * o_lin into o_comb.
// q/k chunk tiles are pure copies -> async global->LDS when available.
// ---------------------------------------------------------------------------
__global__ __launch_bounds__(256) void delta_kernel(
    const float* __restrict__ qlb, const float* __restrict__ klb,
    const float* __restrict__ glb, const float* __restrict__ vbuf,
    float* __restrict__ ocomb) {
  __shared__ __attribute__((aligned(16))) float Sst[64][68];  // state S[d][e]
  __shared__ __attribute__((aligned(16))) float qs[64][68];   // feat q chunk
  __shared__ __attribute__((aligned(16))) float ks[64][68];   // feat k chunk
  __shared__ __attribute__((aligned(16))) float us[64][68];   // v_beta -> u -> u_p
  __shared__ __attribute__((aligned(16))) float wsm[64][68];  // k_beta -> w -> o
  __shared__ __attribute__((aligned(16))) float Ls[64][68];   // L -> attn

  const int tid = threadIdx.x;
  const int bh  = blockIdx.x;
  const int b   = bh >> 5, h = bh & 31, kvh = h >> 2;

  for (int i = tid; i < 64 * 64; i += 256) Sst[i >> 6][i & 63] = 0.f;
  __syncthreads();

  for (int ch = 0; ch < 16; ++ch) {
    const int n0 = ch * 64;

#if USE_ASYNC
    // async copy of q/k chunk tiles: 1024 16-byte transfers each
    for (int i = tid; i < 64 * 16; i += 256) {
      int c = i >> 4, d4 = (i & 15) << 2;
      size_t rq = (size_t)(b * NN + n0 + c) * HIDC + h * DD + d4;
      size_t rk = (size_t)(b * NN + n0 + c) * (HKVC * DD) + kvh * DD + d4;
      __builtin_amdgcn_global_load_async_to_lds_b128(
          ASYNC_PTR(qlb + rq), ASYNC_PTR(&qs[c][d4]), 0, 0);
      __builtin_amdgcn_global_load_async_to_lds_b128(
          ASYNC_PTR(klb + rk), ASYNC_PTR(&ks[c][d4]), 0, 0);
    }
#endif
    float greg[16];
    for (int i = tid, t = 0; i < 64 * 64; i += 256, ++t) {
      int c = i >> 6, d = i & 63;
      size_t rkv = (size_t)(b * NN + n0 + c) * (HKVC * DD) + kvh * DD + d;
      float gv = glb[rkv];
      greg[t] = gv;
      us[c][d] = vbuf[rkv] * gv;  // v_beta
#if !USE_ASYNC
      size_t rq = (size_t)(b * NN + n0 + c) * HIDC + h * DD + d;
      qs[c][d] = qlb[rq];
      ks[c][d] = klb[rkv];
#endif
    }
#if USE_ASYNC
    __builtin_amdgcn_s_wait_asynccnt(0);
#endif
    __syncthreads();
    for (int i = tid, t = 0; i < 64 * 64; i += 256, ++t) {
      int c = i >> 6, d = i & 63;
      wsm[c][d] = ks[c][d] * greg[t];  // k_beta
    }
    __syncthreads();

    // L = k_beta . k^T  (only strict lower triangle is consumed)
    mm64<false, true, 0>(wsm, ks, Ls, 1.f, 0.f);
    __syncthreads();

    // Unit-lower-triangular solve (I + tril(L,-1)) x = rhs, in place, for
    // both u (rhs=v_beta) and w (rhs=k_beta). Columns independent -> no
    // barriers inside; 128 lanes solve both systems.
    if (tid < 128) {
      float (*X)[68] = (tid & 64) ? wsm : us;
      int c = tid & 63;
      for (int i = 1; i < 64; ++i) {
        float a = 0.f;
        for (int j = 0; j < i; ++j) a += Ls[i][j] * X[j][c];
        X[i][c] -= a;
      }
    }
    __syncthreads();

    // u_p = u - w . S
    mm64<false, false, 0>(wsm, Sst, us, -1.f, 1.f);
    __syncthreads();

    // attn = tril(q . k^T)  (diagonal kept)
    mm64<false, true, 1>(qs, ks, Ls, 1.f, 0.f);
    __syncthreads();

    // o = q . S  (into wsm; w is dead now)
    mm64<false, false, 0>(qs, Sst, wsm, 1.f, 0.f);
    __syncthreads();

    // o += attn . u_p
    mm64<false, false, 0>(Ls, us, wsm, 1.f, 1.f);
    __syncthreads();

    // S += k^T . u_p   (overlaps with the o store below)
    mm64<true, false, 0>(ks, us, Sst, 1.f, 1.f);

    for (int i = tid; i < 64 * 64; i += 256) {
      int c = i >> 6, d = i & 63;
      ocomb[(size_t)(b * NN + n0 + c) * HIDC + h * DD + d] = 0.5f * wsm[c][d];
    }
    __syncthreads();
  }
}

// ---------------------------------------------------------------------------
// Host-side orchestration.
// Inputs: 0=hidden_states f32[2,1024,2048], 1=Wq[2048,2048], 2=Wk[2048,512],
//         3=Wv[2048,512], 4=Wo[2048,2048].
// Output: out f32[2,1024,2048] ++ attn f32[2,32,1024,1024].
// Workspace: 16,777,216 floats = 64 MiB.
// ---------------------------------------------------------------------------
extern "C" void kernel_launch(void* const* d_in, const int* in_sizes, int n_in,
                              void* d_out, int out_size, void* d_ws, size_t ws_size,
                              hipStream_t stream) {
  (void)in_sizes; (void)n_in; (void)out_size; (void)ws_size;
  const float* hs = (const float*)d_in[0];
  const float* Wq = (const float*)d_in[1];
  const float* Wk = (const float*)d_in[2];
  const float* Wv = (const float*)d_in[3];
  const float* Wo = (const float*)d_in[4];

  float* out  = (float*)d_out;
  float* attn = out + (size_t)BB * NN * HIDC;   // second tuple output

  float* wsf   = (float*)d_ws;
  float* qbuf  = wsf;                    // 4,194,304
  float* kbuf  = qbuf + (size_t)4194304; // 1,048,576
  float* vbuf  = kbuf + (size_t)1048576; // 1,048,576
  float* qlb   = vbuf + (size_t)1048576; // 4,194,304
  float* klb   = qlb  + (size_t)4194304; // 1,048,576
  float* glb   = klb  + (size_t)1048576; // 1,048,576
  float* ocomb = glb  + (size_t)1048576; // 4,194,304

  const int Mrows = BB * NN;  // 2048

  // 1) QKV projections (bf16 WMMA GEMMs)
  gemm_nn_kernel<<<dim3(HIDC / 64, Mrows / 16), 128, 0, stream>>>(
      hs, HIDC, Wq, HIDC, qbuf, HIDC, HIDC, 1.f, 0.f);
  gemm_nn_kernel<<<dim3((HKVC * DD) / 64, Mrows / 16), 128, 0, stream>>>(
      hs, HIDC, Wk, HKVC * DD, kbuf, HKVC * DD, HIDC, 1.f, 0.f);
  gemm_nn_kernel<<<dim3((HKVC * DD) / 64, Mrows / 16), 128, 0, stream>>>(
      hs, HIDC, Wv, HKVC * DD, vbuf, HKVC * DD, HIDC, 1.f, 0.f);

  // 2) feature maps: ql = softmax(q), kl = softmax(k), gl = logsigmoid(k)/16
  featmap_kernel<<<(BB * NN * HH) / 8, 256, 0, stream>>>(qbuf, qlb, nullptr);
  featmap_kernel<<<(BB * NN * HKVC) / 8, 256, 0, stream>>>(kbuf, klb, glb);

  // 3) DeltaNet chunked scan -> ocomb = MAG * o_lin (fills every element)
  delta_kernel<<<BB * HH, 256, 0, stream>>>(qlb, klb, glb, vbuf, ocomb);

  // 4) base branch: masked scores -> softmax (in place, 2nd output) -> P.V
  scores_kernel<<<dim3(NN / 64, NN / 16, BB * HH), 128, 0, stream>>>(qbuf, kbuf, attn);
  softmax_rows<<<BB * HH * NN, 256, 0, stream>>>(attn);
  obase_kernel<<<dim3(1, NN / 16, BB * HH), 128, 0, stream>>>(attn, vbuf, ocomb);

  // 5) out = (0.5*o_lin + 0.5*o_base) @ Wo
  gemm_nn_kernel<<<dim3(HIDC / 64, Mrows / 16), 128, 0, stream>>>(
      ocomb, HIDC, Wo, HIDC, out, HIDC, HIDC, 1.f, 0.f);
}
